// Single_CGCNN_19009525252277
// MI455X (gfx1250) — matrix operand order repaired and checked by reading, not compiled, not run on hardware
//
#include <hip/hip_runtime.h>

// ---------------------------------------------------------------------------
// CGCNN forward for MI455X (gfx1250, wave32, WMMA bf16 16x16x32)
// Dominant cost: per-layer gated GEMM (786432 x 297) @ (297 x 256), done twice
// per layer (stats pass + apply pass) to implement train-mode BatchNorm
// without materializing the 400MB 'gated' tensor.
// Round 3: rolled ct-tile loops (#pragma unroll 1) + __launch_bounds__(256,1)
// to eliminate the accumulator scratch spills seen in rounds 1-2.
// ---------------------------------------------------------------------------

#define NATOMS   65536
#define MNBR     12
#define CFEA     128      // hidden atom feature
#define KBOND    41
#define KTOT     297      // 2C + K
#define KPAD     320      // padded to 10 chunks of 32
#define NKCH     10       // K chunks of 32
#define OTOT     256      // 2C output channels
#define NCT      16       // 256/16 col tiles
#define NROWS    786432   // NATOMS * MNBR
#define ROWS_PER_BLK 384  // 8 waves * 3 tiles * 16 rows = 32 atoms
#define NBLK_GEMM 2048    // NROWS / ROWS_PER_BLK
#define N0CRYS   512
#define HFC      256

typedef __attribute__((ext_vector_type(16))) __bf16          v16bf;
typedef __attribute__((ext_vector_type(16))) unsigned short  v16us;
typedef __attribute__((ext_vector_type(8)))  float           v8f;

__device__ __forceinline__ unsigned short f2bf(float f) {
    unsigned u = __builtin_bit_cast(unsigned, f);
    unsigned r = u + 0x7FFFu + ((u >> 16) & 1u);   // round-to-nearest-even
    return (unsigned short)(r >> 16);
}
__device__ __forceinline__ float softplus_f(float x) {
    return (x > 20.f) ? x : log1pf(expf(x));
}
__device__ __forceinline__ float sigmoid_f(float x) {
    return 1.f / (1.f + expf(-x));
}

// Build WMMA bf16 A fragment (16x32 tile) from an LDS-staged 16x320 row tile.
// 16-bit A layout: lanes 0-15 row M=lane, k {0..7,16..23}; lanes 16-31 row
// M=lane-16, k {8..15,24..31}.
__device__ __forceinline__ v16bf load_afrag(const unsigned short* ldsTile,
                                            int lane, int kc) {
    const int row = lane & 15;
    const int off = (lane >> 4) << 3;                 // 0 or 8
    const unsigned short* p = ldsTile + row * KPAD + kc * 32;
    v16us t;
#pragma unroll
    for (int e = 0; e < 8; ++e) {
        t[e]     = p[off + e];
        t[e + 8] = p[off + 16 + e];
    }
    return __builtin_bit_cast(v16bf, t);
}

__device__ __forceinline__ v16bf load_bfrag(const unsigned short* ldsB,
                                            int kc, int ct, int lane) {
    const v16us* bp = (const v16us*)ldsB + ((kc * NCT + ct) * 32 + lane);
    return __builtin_bit_cast(v16bf, *bp);
}

// Stage one 16-row x 320-col bf16 tile of "total" = [self | nbr | bond | 0pad]
__device__ __forceinline__ void stage_tileA(unsigned short* ldsTile,
                                            const float* __restrict__ atom,
                                            const float* __restrict__ nbrF,
                                            const int* __restrict__ nbrIdx,
                                            int rowBase, int lane) {
    for (int row = 0; row < 16; ++row) {
        int r = rowBase + row;
        int n = r / MNBR;
        int m = r - n * MNBR;
        int j = nbrIdx[n * MNBR + m];
        const float* an = atom + (long)n * CFEA;
        const float* aj = atom + (long)j * CFEA;
        const float* nf = nbrF + ((long)n * MNBR + m) * KBOND;
        unsigned short* dst = ldsTile + row * KPAD;
        for (int k = lane; k < KPAD; k += 32) {
            float v;
            if      (k < CFEA)       v = an[k];
            else if (k < 2 * CFEA)   v = aj[k - CFEA];
            else if (k < KTOT)       v = nf[k - 2 * CFEA];
            else                     v = 0.f;
            dst[k] = f2bf(v);
        }
    }
}

// ---------------------------------------------------------------------------
// One-time: reformat conv_W (L,256,297) f32 -> bf16 fragments in the exact
// WMMA B layout: Bbuf[l][kc][ct][lane][e], K = kc*32 + (lane<16 ? 0 : 8) +
// (e<8 ? e : 8+e), N = ct*16 + lane%16.
// ---------------------------------------------------------------------------
__global__ void format_W_kernel(const float* __restrict__ convW,
                                unsigned short* __restrict__ Wbuf) {
    int t = blockIdx.x * blockDim.x + threadIdx.x;
    if (t >= 4 * NKCH * NCT * 32) return;
    int lane = t & 31;
    int rest = t >> 5;
    int ct = rest & 15;  rest >>= 4;
    int kc = rest % NKCH;
    int l  = rest / NKCH;
    int o     = ct * 16 + (lane & 15);
    int kbase = kc * 32 + ((lane >> 4) ? 8 : 0);
    unsigned short* dst = Wbuf + (((long)l * NKCH * NCT + kc * NCT + ct) * 32 + lane) * 16;
    const float* src = convW + ((long)l * OTOT + o) * KTOT;
#pragma unroll
    for (int e = 0; e < 16; ++e) {
        int k = kbase + (e < 8 ? e : 8 + e);
        dst[e] = f2bf((k < KTOT) ? src[k] : 0.f);
    }
}

// ---------------------------------------------------------------------------
// Embedding: atom = atom_fea (N x 92) @ W_emb^T (92->128) + b
// ---------------------------------------------------------------------------
__global__ void embed_kernel(const float* __restrict__ af,
                             const float* __restrict__ W,
                             const float* __restrict__ b,
                             float* __restrict__ out) {
    __shared__ float row[92];
    int n = blockIdx.x;
    int c = threadIdx.x;
    if (c < 92) row[c] = af[(long)n * 92 + c];
    __syncthreads();
    const float* w = W + (long)c * 92;
    float s = b[c];
#pragma unroll 4
    for (int k = 0; k < 92; ++k) s += row[k] * w[k];
    out[(long)n * CFEA + c] = s;
}

// ---------------------------------------------------------------------------
// Pass A: WMMA GEMM, accumulate per-channel sum / sumsq of gated (incl. bias)
// ---------------------------------------------------------------------------
__global__ void __launch_bounds__(256, 1)
conv_stats_kernel(const float* __restrict__ atom,
                  const float* __restrict__ nbrF,
                  const int* __restrict__ nbrIdx,
                  const unsigned short* __restrict__ Wbuf,
                  const float* __restrict__ convb,
                  float* __restrict__ gsum,
                  float* __restrict__ gsq) {
    extern __shared__ unsigned char smem[];
    unsigned short* ldsB = (unsigned short*)smem;                    // 163840 B
    unsigned short* ldsA = (unsigned short*)(smem + 163840);         //  81920 B
    float* sSum = (float*)(smem + 163840 + 81920);                   //   1024 B
    float* sSq  = sSum + OTOT;                                       //   1024 B

    const int tid  = threadIdx.x;
    const int lane = tid & 31;
    const int wave = tid >> 5;

    {   // stage W fragments into LDS (uint4 copies)
        const uint4* src = (const uint4*)Wbuf;
        uint4* dst = (uint4*)ldsB;
        for (int i = tid; i < 163840 / 16; i += 256) dst[i] = src[i];
    }
    sSum[tid] = 0.f; sSq[tid] = 0.f;    // blockDim == 256 == OTOT
    __syncthreads();

    unsigned short* myA = ldsA + wave * 16 * KPAD;
    const int waveRow0 = blockIdx.x * ROWS_PER_BLK + wave * 48;

    for (int t = 0; t < 3; ++t) {
        const int rowBase = waveRow0 + t * 16;
        stage_tileA(myA, atom, nbrF, nbrIdx, rowBase, lane);
        v16bf af[NKCH];
#pragma unroll
        for (int kc = 0; kc < NKCH; ++kc) af[kc] = load_afrag(myA, lane, kc);

#pragma unroll 1          // keep ONE accumulator chain live -> no spills
        for (int ct = 0; ct < NCT; ++ct) {
            v8f acc = {0.f, 0.f, 0.f, 0.f, 0.f, 0.f, 0.f, 0.f};
#pragma unroll
            for (int kc = 0; kc < NKCH; ++kc) {
                acc = __builtin_amdgcn_wmma_f32_16x16x32_bf16(
                    false, af[kc], false, load_bfrag(ldsB, kc, ct, lane),
                    (short)0, acc, false, false);
            }
            const int col = ct * 16 + (lane & 15);
            const float bb = convb[col];
            float s = 0.f, q = 0.f;
#pragma unroll
            for (int i = 0; i < 8; ++i) {
                float g = acc[i] + bb;
                s += g; q += g * g;
            }
            atomicAdd(&sSum[col], s);
            atomicAdd(&sSq[col], q);
        }
    }
    __syncthreads();
    atomicAdd(&gsum[tid], sSum[tid]);
    atomicAdd(&gsq[tid],  sSq[tid]);
}

// ---------------------------------------------------------------------------
// Fold (sum, sumsq, gamma, beta) -> affine (a, c): norm = a*gated + c
// ---------------------------------------------------------------------------
__global__ void bn_params_kernel(const float* __restrict__ sum,
                                 const float* __restrict__ sq,
                                 const float* __restrict__ g,
                                 const float* __restrict__ b,
                                 float invR, int n,
                                 float* __restrict__ a_out,
                                 float* __restrict__ c_out) {
    int i = blockIdx.x * blockDim.x + threadIdx.x;
    if (i >= n) return;
    float m = sum[i] * invR;
    float v = sq[i] * invR - m * m;
    float a = g[i] * rsqrtf(v + 1e-5f);
    a_out[i] = a;
    c_out[i] = b[i] - m * a;
}

// ---------------------------------------------------------------------------
// Pass B: recompute GEMM; per iteration compute PAIRED column tiles
// (filter ct, core ct+8) with two accumulators. Apply bn1, sigmoid*softplus,
// reduce over M into per-atom summed, accumulate bn2 stats.
// ---------------------------------------------------------------------------
__global__ void __launch_bounds__(256, 1)
conv_apply_kernel(const float* __restrict__ atom,
                  const float* __restrict__ nbrF,
                  const int* __restrict__ nbrIdx,
                  const unsigned short* __restrict__ Wbuf,
                  const float* __restrict__ convb,
                  const float* __restrict__ a1,
                  const float* __restrict__ c1,
                  float* __restrict__ summedOut,
                  float* __restrict__ g2sum,
                  float* __restrict__ g2sq) {
    extern __shared__ unsigned char smem[];
    unsigned short* ldsB = (unsigned short*)smem;                    // 163840 B
    unsigned short* ldsA = (unsigned short*)(smem + 163840);         //  81920 B
    float* ldsSum = (float*)(smem + 245760);                         //  16384 B
    float* s2s    = (float*)(smem + 262144);                         //    512 B
    float* s2q    = (float*)(smem + 262656);                         //    512 B

    const int tid  = threadIdx.x;
    const int lane = tid & 31;
    const int wave = tid >> 5;

    {
        const uint4* src = (const uint4*)Wbuf;
        uint4* dst = (uint4*)ldsB;
        for (int i = tid; i < 163840 / 16; i += 256) dst[i] = src[i];
    }
    for (int i = tid; i < 32 * CFEA; i += 256) ldsSum[i] = 0.f;
    if (tid < CFEA) { s2s[tid] = 0.f; s2q[tid] = 0.f; }
    __syncthreads();

    unsigned short* myA = ldsA + wave * 16 * KPAD;
    const int waveRow0 = blockIdx.x * ROWS_PER_BLK + wave * 48;
    const int atomBase = blockIdx.x * 32;
    const int half8    = (lane >> 4) << 3;   // 0 or 8 row offset within tile

    for (int t = 0; t < 3; ++t) {
        const int rowBase = waveRow0 + t * 16;
        stage_tileA(myA, atom, nbrF, nbrIdx, rowBase, lane);
        v16bf af[NKCH];
#pragma unroll
        for (int kc = 0; kc < NKCH; ++kc) af[kc] = load_afrag(myA, lane, kc);

#pragma unroll 1          // paired tiles (ct, ct+8); exactly TWO acc chains live
        for (int ct = 0; ct < 8; ++ct) {
            v8f accF = {0.f, 0.f, 0.f, 0.f, 0.f, 0.f, 0.f, 0.f};
            v8f accC = {0.f, 0.f, 0.f, 0.f, 0.f, 0.f, 0.f, 0.f};
#pragma unroll
            for (int kc = 0; kc < NKCH; ++kc) {
                accF = __builtin_amdgcn_wmma_f32_16x16x32_bf16(
                    false, af[kc], false, load_bfrag(ldsB, kc, ct, lane),
                    (short)0, accF, false, false);
            }
#pragma unroll
            for (int kc = 0; kc < NKCH; ++kc) {
                accC = __builtin_amdgcn_wmma_f32_16x16x32_bf16(
                    false, af[kc], false, load_bfrag(ldsB, kc, ct + 8, lane),
                    (short)0, accC, false, false);
            }
            const int colF = ct * 16 + (lane & 15);        // filter col (< 128)
            const int colC = colF + CFEA;                  // core col
            const float aaF = a1[colF], ccF = c1[colF], bbF = convb[colF];
            const float aaC = a1[colC], ccC = c1[colC], bbC = convb[colC];
#pragma unroll
            for (int i = 0; i < 8; ++i) {
                float f = sigmoid_f(aaF * (accF[i] + bbF) + ccF);
                float c = softplus_f(aaC * (accC[i] + bbC) + ccC);
                float p = f * c;
                int r = rowBase + i + half8;
                int localAtom = r / MNBR - atomBase;
                atomicAdd(&ldsSum[localAtom * CFEA + colF], p);
            }
        }
    }
    __syncthreads();

    // flush summed + bn2 stats
    for (int idx = tid; idx < 32 * CFEA; idx += 256) {
        float v = ldsSum[idx];
        summedOut[(long)atomBase * CFEA + idx] = v;
        int c = idx & (CFEA - 1);
        atomicAdd(&s2s[c], v);
        atomicAdd(&s2q[c], v * v);
    }
    __syncthreads();
    if (tid < CFEA) {
        atomicAdd(&g2sum[tid], s2s[tid]);
        atomicAdd(&g2sq[tid],  s2q[tid]);
    }
}

// ---------------------------------------------------------------------------
// atom_new = softplus(atom + bn2(summed))
// ---------------------------------------------------------------------------
__global__ void update_atoms_kernel(const float* __restrict__ atomIn,
                                    const float* __restrict__ summed,
                                    const float* __restrict__ a2,
                                    const float* __restrict__ c2,
                                    float* __restrict__ atomOut) {
    long idx = (long)blockIdx.x * blockDim.x + threadIdx.x;
    if (idx >= (long)NATOMS * CFEA) return;
    int c = (int)(idx & (CFEA - 1));
    float x = atomIn[idx] + summed[idx] * a2[c] + c2[c];
    atomOut[idx] = softplus_f(x);
}

__global__ void zero_kernel(float* p, long n) {
    long i = (long)blockIdx.x * blockDim.x + threadIdx.x;
    if (i < n) p[i] = 0.f;
}

// ---------------------------------------------------------------------------
// Pooling: per-crystal mean (atomic segment-sum) then softplus
// ---------------------------------------------------------------------------
__global__ void pool_acc_kernel(const float* __restrict__ atom,
                                const int* __restrict__ cryIdx,
                                float* __restrict__ crysSum,
                                float* __restrict__ cnt) {
    long idx = (long)blockIdx.x * blockDim.x + threadIdx.x;
    if (idx >= (long)NATOMS * CFEA) return;
    int n = (int)(idx >> 7);
    int c = (int)(idx & (CFEA - 1));
    int cid = cryIdx[n];
    atomicAdd(&crysSum[(long)cid * CFEA + c], atom[idx]);
    if (c == 0) atomicAdd(&cnt[cid], 1.f);
}

__global__ void pool_fin_kernel(const float* __restrict__ crysSum,
                                const float* __restrict__ cnt,
                                float* __restrict__ crysFeat) {
    int idx = blockIdx.x * blockDim.x + threadIdx.x;
    if (idx >= N0CRYS * CFEA) return;
    crysFeat[idx] = softplus_f(crysSum[idx] / cnt[idx >> 7]);
}

// ---------------------------------------------------------------------------
// Small FC: out[i][o] = softplus( in[i][:] . W[o][:] + b[o] )
// ---------------------------------------------------------------------------
__global__ void fc_kernel(const float* __restrict__ in,
                          const float* __restrict__ W,
                          const float* __restrict__ b,
                          float* __restrict__ out, int Kd, int Od) {
    int t = blockIdx.x * blockDim.x + threadIdx.x;
    if (t >= N0CRYS * Od) return;
    int i = t / Od, o = t - i * Od;
    const float* x = in + (long)i * Kd;
    const float* w = W + (long)o * Kd;
    float s = b[o];
    for (int k = 0; k < Kd; ++k) s += x[k] * w[k];
    out[t] = softplus_f(s);
}

__global__ void out_kernel(const float* __restrict__ h,
                           const float* __restrict__ W,
                           const float* __restrict__ b,
                           float* __restrict__ out) {
    int t = blockIdx.x * blockDim.x + threadIdx.x;
    if (t >= N0CRYS) return;
    const float* x = h + (long)t * HFC;
    float s = b[0];
    for (int k = 0; k < HFC; ++k) s += x[k] * W[k];
    out[t] = s;
}

// ---------------------------------------------------------------------------
extern "C" void kernel_launch(void* const* d_in, const int* in_sizes, int n_in,
                              void* d_out, int out_size, void* d_ws, size_t ws_size,
                              hipStream_t stream) {
    const float* atom_fea = (const float*)d_in[0];
    const float* nbr_fea  = (const float*)d_in[1];
    const int*   nbr_idx  = (const int*)d_in[2];
    const int*   cry_idx  = (const int*)d_in[3];
    const float* W_emb    = (const float*)d_in[4];
    const float* b_emb    = (const float*)d_in[5];
    const float* conv_W   = (const float*)d_in[6];
    const float* conv_b   = (const float*)d_in[7];
    const float* bn1_g    = (const float*)d_in[8];
    const float* bn1_b    = (const float*)d_in[9];
    const float* bn2_g    = (const float*)d_in[10];
    const float* bn2_b    = (const float*)d_in[11];
    const float* fc1_W    = (const float*)d_in[12];
    const float* fc1_b    = (const float*)d_in[13];
    const float* fch_W    = (const float*)d_in[14];
    const float* fch_b    = (const float*)d_in[15];
    const float* out_W    = (const float*)d_in[16];
    const float* out_b    = (const float*)d_in[17];
    float* out = (float*)d_out;

    // ---- workspace carve-out ----
    char* ws = (char*)d_ws;
    size_t off = 0;
    auto carve = [&](size_t bytes) -> void* {
        void* p = ws + off;
        off += (bytes + 255) & ~(size_t)255;
        return p;
    };
    unsigned short* Wbuf = (unsigned short*)carve((size_t)4 * NKCH * NCT * 32 * 16 * 2); // 640 KB
    float* atomA  = (float*)carve((size_t)NATOMS * CFEA * 4);
    float* atomB  = (float*)carve((size_t)NATOMS * CFEA * 4);
    float* summed = (float*)carve((size_t)NATOMS * CFEA * 4);
    float* stats  = (float*)carve((size_t)(768 + 768) * 4); // sums + affine params
    float* sum1 = stats;            // 256
    float* sq1  = stats + 256;      // 256
    float* sum2 = stats + 512;      // 128
    float* sq2  = stats + 640;      // 128  (first 768 floats zeroed per layer)
    float* a1   = stats + 768;      // 256
    float* c1   = stats + 1024;     // 256
    float* a2   = stats + 1280;     // 128
    float* c2   = stats + 1408;     // 128
    float* crysSum  = (float*)carve((size_t)(N0CRYS * CFEA + N0CRYS) * 4);
    float* cnt      = crysSum + N0CRYS * CFEA;
    float* crysFeat = (float*)carve((size_t)N0CRYS * CFEA * 4);
    float* h1       = (float*)carve((size_t)N0CRYS * HFC * 4);
    float* h2       = (float*)carve((size_t)N0CRYS * HFC * 4);
    (void)ws_size; (void)in_sizes; (void)n_in; (void)out_size;

    const size_t smemA = 163840 + 81920 + 2048;            // stats pass
    const size_t smemB = 163840 + 81920 + 16384 + 1024;    // apply pass

    // W -> bf16 fragment layout (once per launch)
    format_W_kernel<<<(4 * NKCH * NCT * 32 + 255) / 256, 256, 0, stream>>>(conv_W, Wbuf);

    // embedding
    embed_kernel<<<NATOMS, 128, 0, stream>>>(atom_fea, W_emb, b_emb, atomA);

    float* cur = atomA;
    float* nxt = atomB;
    for (int l = 0; l < 4; ++l) {
        const unsigned short* Wl = Wbuf + (size_t)l * NKCH * NCT * 32 * 16;
        const float* cb = conv_b + l * OTOT;

        zero_kernel<<<3, 256, 0, stream>>>(stats, 768);

        conv_stats_kernel<<<NBLK_GEMM, 256, smemA, stream>>>(
            cur, nbr_fea, nbr_idx, Wl, cb, sum1, sq1);

        bn_params_kernel<<<1, 256, 0, stream>>>(
            sum1, sq1, bn1_g + l * OTOT, bn1_b + l * OTOT,
            1.0f / (float)NROWS, OTOT, a1, c1);

        conv_apply_kernel<<<NBLK_GEMM, 256, smemB, stream>>>(
            cur, nbr_fea, nbr_idx, Wl, cb, a1, c1, summed, sum2, sq2);

        bn_params_kernel<<<1, 128, 0, stream>>>(
            sum2, sq2, bn2_g + l * CFEA, bn2_b + l * CFEA,
            1.0f / (float)NATOMS, CFEA, a2, c2);

        update_atoms_kernel<<<(NATOMS * CFEA) / 256, 256, 0, stream>>>(
            cur, summed, a2, c2, nxt);

        float* tmp = cur; cur = nxt; nxt = tmp;
    }

    // pooling
    zero_kernel<<<(N0CRYS * CFEA + N0CRYS + 255) / 256, 256, 0, stream>>>(
        crysSum, N0CRYS * CFEA + N0CRYS);
    pool_acc_kernel<<<(NATOMS * CFEA) / 256, 256, 0, stream>>>(cur, cry_idx, crysSum, cnt);
    pool_fin_kernel<<<(N0CRYS * CFEA) / 256, 256, 0, stream>>>(crysSum, cnt, crysFeat);

    // MLP head
    fc_kernel<<<(N0CRYS * HFC) / 256, 256, 0, stream>>>(crysFeat, fc1_W, fc1_b, h1, CFEA, HFC);
    fc_kernel<<<(N0CRYS * HFC) / 256, 256, 0, stream>>>(h1, fch_W, fch_b, h2, HFC, HFC);
    out_kernel<<<2, 256, 0, stream>>>(h2, out_W, out_b, out);
}
